// LocalSpatialAttentionChunk_58119497450442
// MI455X (gfx1250) — matrix operand order repaired and checked by reading, not compiled, not run on hardware
//
#include <hip/hip_runtime.h>
#include <hip/hip_bf16.h>

typedef __attribute__((ext_vector_type(16))) _Float16 v16h;
typedef __attribute__((ext_vector_type(8)))  float    v8f;

struct __align__(16) B16 { unsigned int w[4]; };

union Frag {
    v16h v;
    _Float16 e[16];
    unsigned u[8];
    B16 q4[2];
};
union CFrag {
    v8f v;
    float f[8];
};

__device__ __forceinline__ int clampi(int v, int lo, int hi) {
    return v < lo ? lo : (v > hi ? hi : v);
}

// Async 16B copy global -> LDS (CDNA5 GLOBAL_LOAD_ASYNC_TO_LDS_B128).
// VDST carries the LDS byte offset == low 32 bits of the generic LDS pointer.
__device__ __forceinline__ void async_copy_b128(void* lds_dst, const void* gsrc) {
    asm volatile("global_load_async_to_lds_b128 %0, %1, off"
                 :: "v"((unsigned)(size_t)lds_dst),
                    "v"((unsigned long long)(size_t)gsrc)
                 : "memory");
}
#define WAIT_ASYNCCNT_3() asm volatile("s_wait_asynccnt 0x3" ::: "memory")
#define WAIT_ASYNCCNT_0() asm volatile("s_wait_asynccnt 0x0" ::: "memory")

// ---------------------------------------------------------------------------
// Kernel 1: convert inputs to f16, pre-transpose weights so WMMA B-fragments
// are contiguous.  WqkvT layout: [n (0..767)][k (0..255)], WoT: [n][k].
// ---------------------------------------------------------------------------
#define NX_ELEMS (16384 * 256)
#define NWQKV    (768 * 256)
#define NWO      (256 * 256)
#define NPOS     (49 * 256)

__global__ __launch_bounds__(256) void lsa_convert_pack(
    const float* __restrict__ x,  const float* __restrict__ Wq,
    const float* __restrict__ Wk, const float* __restrict__ Wv,
    const float* __restrict__ Wo, const float* __restrict__ pos,
    _Float16* __restrict__ xh, _Float16* __restrict__ WqkvT,
    _Float16* __restrict__ WoT, _Float16* __restrict__ posh)
{
    int idx = blockIdx.x * 256 + threadIdx.x;
    if (idx < NX_ELEMS) { xh[idx] = (_Float16)x[idx]; return; }
    idx -= NX_ELEMS;
    if (idx < NWQKV) {
        int n = idx >> 8, k = idx & 255;
        float v;
        if (n < 256)      v = Wq[k * 256 + n];
        else if (n < 512) v = Wk[k * 256 + (n - 256)];
        else              v = Wv[k * 256 + (n - 512)];
        WqkvT[idx] = (_Float16)v;
        return;
    }
    idx -= NWQKV;
    if (idx < NWO) {
        int n = idx >> 8, k = idx & 255;
        WoT[idx] = (_Float16)Wo[k * 256 + n];
        return;
    }
    idx -= NWO;
    if (idx < NPOS) posh[idx] = (_Float16)pos[idx];
}

// ---------------------------------------------------------------------------
// Kernel 2/4: WMMA GEMM, 256 threads = 8 waves.  Block tile 128(M) x 64(N),
// K = 256 in steps of 32.  Double-buffered LDS tiles filled with
// GLOBAL_LOAD_ASYNC_TO_LDS_B128; ASYNCcnt-pipelined (3 async ops per wave
// per tile, in-order completion -> s_wait_asynccnt 3 retires the live tile).
// mode 0: QKV projection (N=768).  Q gets pos_emb[qidx] fused; f16 outputs.
// mode 1: output projection (N=256), f32 output to d_out.
// ---------------------------------------------------------------------------
__global__ __launch_bounds__(256) void lsa_wmma_gemm(
    const _Float16* __restrict__ A, const _Float16* __restrict__ BT,
    _Float16* __restrict__ Qh, _Float16* __restrict__ Kh,
    _Float16* __restrict__ Vh, const float* __restrict__ pos,
    float* __restrict__ outf, int N, int mode)
{
    __shared__ __align__(16) _Float16 At[2][128][32];
    __shared__ __align__(16) _Float16 Bt[2][64][32];

    const int tid = threadIdx.x;
    const int nblk = N >> 6;
    const int bm = blockIdx.x / nblk, bn = blockIdx.x % nblk;
    const int m0 = bm << 7, n0 = bn << 6;
    const int w = tid >> 5, L = tid & 31, hi = L >> 4, lm = L & 15;

    // per-thread fixed tile-copy assignment (3 x b128 per thread per tile)
    const int arow = tid >> 1, akof = (tid & 1) * 16;
    const int bnrow = tid >> 2, bkof = (tid & 3) * 8;
    const _Float16* gA = A + (size_t)(m0 + arow) * 256 + akof;
    const _Float16* gB = BT + (size_t)(n0 + bnrow) * 256 + bkof;

    CFrag acc[4];
    for (int t = 0; t < 4; ++t)
        for (int i = 0; i < 8; ++i) acc[t].f[i] = 0.0f;

    // prologue: issue tile 0
    async_copy_b128(&At[0][arow][akof],      gA);
    async_copy_b128(&At[0][arow][akof + 8],  gA + 8);
    async_copy_b128(&Bt[0][bnrow][bkof],     gB);

    for (int it = 0; it < 8; ++it) {
        const int buf = it & 1;
        if (it < 7) {   // issue tile it+1 into the other buffer
            const int k1 = (it + 1) * 32;
            async_copy_b128(&At[buf ^ 1][arow][akof],     gA + k1);
            async_copy_b128(&At[buf ^ 1][arow][akof + 8], gA + k1 + 8);
            async_copy_b128(&Bt[buf ^ 1][bnrow][bkof],    gB + k1);
            WAIT_ASYNCCNT_3();   // tile `it` complete (in-order), next in flight
        } else {
            WAIT_ASYNCCNT_0();
        }
        __syncthreads();

        Frag a;
        const int afr = w * 16 + lm;
        for (int v = 0; v < 8; ++v) {
            int kk = 2 * v + hi * 16;
            a.u[v] = *(const unsigned*)&At[buf][afr][kk];
        }
        for (int t = 0; t < 4; ++t) {
            Frag b;
            const _Float16* bp = &Bt[buf][t * 16 + lm][hi * 16];
            b.q4[0] = ((const B16*)bp)[0];
            b.q4[1] = ((const B16*)bp)[1];
            acc[t].v = __builtin_amdgcn_wmma_f32_16x16x32_f16(
                false, a.v, false, b.v, (short)0, acc[t].v, false, false);
        }
        __syncthreads();   // all waves done reading before buf is overwritten
    }

    // epilogue
    for (int t = 0; t < 4; ++t) {
        for (int r = 0; r < 8; ++r) {
            int row = m0 + w * 16 + r + hi * 8;
            int col = n0 + t * 16 + lm;
            float val = acc[t].f[r];
            if (mode == 0) {
                if (col < 256) {
                    int y = (row >> 6) & 63, xx = row & 63;
                    int qy = y - clampi(y, 3, 60) + 3;
                    int qx = xx - clampi(xx, 3, 60) + 3;
                    val += pos[(qy * 7 + qx) * 256 + col];
                    Qh[(size_t)row * 256 + col] = (_Float16)val;
                } else if (col < 512) {
                    Kh[(size_t)row * 256 + (col - 256)] = (_Float16)val;
                } else {
                    Vh[(size_t)row * 256 + (col - 512)] = (_Float16)val;
                }
            } else {
                outf[(size_t)row * 256 + col] = val;
            }
        }
    }
}

// ---------------------------------------------------------------------------
// Kernel 3: windowed flash attention.  1 block = 16 queries (one y row,
// 16 consecutive x) x 8 heads; 1 wave per head.  Per dy: QK over the <=22
// physical key columns (2 WMMAs), diagonal gather + online softmax via LDS,
// P@V (2 WMMAs).  Q.pos term precomputed as S49 via 4 WMMAs.  V tile is
// staged in LDS with async-to-LDS copies.
// ---------------------------------------------------------------------------
__global__ __launch_bounds__(256) void lsa_attention(
    const _Float16* __restrict__ Qh, const _Float16* __restrict__ Kh,
    const _Float16* __restrict__ Vh, const _Float16* __restrict__ posh,
    _Float16* __restrict__ Oh)
{
    __shared__ __align__(16) _Float16 Ptile[8][16][32];   //  8 KB
    __shared__ __align__(16) float    Rtile[8][16][32];   // 16 KB
    __shared__ __align__(16) _Float16 S49t[8][16][64];    // 16 KB
    __shared__ __align__(16) _Float16 Vtile[32][256];     // 16 KB
    __shared__ float mbuf[8][16], lbuf[8][16], abuf[8][16];

    const int tid = threadIdx.x;
    const int h = tid >> 5, L = tid & 31, hi = L >> 4, lm = L & 15;
    const int bid = blockIdx.x;
    const int xblk = bid & 3, y = (bid >> 2) & 63, b = bid >> 8;
    const int x0 = xblk << 4;
    const int cy = clampi(y, 3, 60);
    const int kxmin = clampi(x0, 3, 60) - 3;
    const int kxmax = clampi(x0 + 15, 3, 60) + 3;
    const int width = kxmax - kxmin + 1;     // 19 or 22, <= 32
    const int base_row = (b * 64 + y) * 64 + x0;
    const float scale = 0.17677669529663687f;

    // ---- Q A-fragment (16 queries x 32 F dims), kept in registers ----
    Frag aq;
    {
        const _Float16* qrow = Qh + (size_t)(base_row + lm) * 256 + h * 32;
        for (int v = 0; v < 8; ++v) {
            int kk = 2 * v + hi * 16;
            aq.u[v] = *(const unsigned*)(qrow + kk);
        }
    }

    // ---- zero Vtile rows >= width (never rewritten; padding for P@V) ----
    for (int i = tid; i < (32 - width) * 32; i += 256) {
        int rrow = width + (i >> 5), c = i & 31;
        B16 z; z.w[0] = z.w[1] = z.w[2] = z.w[3] = 0;
        ((B16*)&Vtile[rrow][0])[c] = z;
    }

    // ---- S49[q][s] = Qa . pos_emb[s]   (pad S 49 -> 64) ----
    {
        CFrag s[4];
        for (int t = 0; t < 4; ++t)
            for (int i = 0; i < 8; ++i) s[t].f[i] = 0.0f;
        for (int t = 0; t < 4; ++t) {
            Frag bp;
            int sidx = t * 16 + lm;
            if (sidx < 49) {
                const _Float16* pp = posh + (size_t)sidx * 256 + h * 32 + hi * 16;
                bp.q4[0] = ((const B16*)pp)[0];
                bp.q4[1] = ((const B16*)pp)[1];
            } else {
                bp.q4[0].w[0] = bp.q4[0].w[1] = bp.q4[0].w[2] = bp.q4[0].w[3] = 0;
                bp.q4[1] = bp.q4[0];
            }
            s[t].v = __builtin_amdgcn_wmma_f32_16x16x32_f16(
                false, aq.v, false, bp.v, (short)0, s[t].v, false, false);
        }
        for (int t = 0; t < 4; ++t)
            for (int r = 0; r < 8; ++r)
                S49t[h][r + hi * 8][t * 16 + lm] = (_Float16)s[t].f[r];
    }
    if (L < 16) { mbuf[h][L] = -1e9f; lbuf[h][L] = 0.0f; }

    CFrag acc0, acc1;
    for (int i = 0; i < 8; ++i) { acc0.f[i] = 0.0f; acc1.f[i] = 0.0f; }

    for (int dy = -3; dy <= 3; ++dy) {
        const int ky = cy + dy;                    // in [0,63]
        const int krowbase = (b * 64 + ky) * 64;

        __syncthreads();    // previous iteration's P@V done before V reload
        // ---- async V tile stage: rows = physical key columns ----
        for (int j = tid >> 3; j < width; j += 32) {
            const _Float16* src = Vh + (size_t)(krowbase + kxmin + j) * 256
                                  + (tid & 7) * 32;
            _Float16* dst = &Vtile[j][(tid & 7) * 32];
            async_copy_b128(dst,      src);
            async_copy_b128(dst + 8,  src + 8);
            async_copy_b128(dst + 16, src + 16);
            async_copy_b128(dst + 24, src + 24);
        }
        WAIT_ASYNCCNT_0();
        __syncthreads();

        // ---- raw scores Qa . K_phys over 32 (padded) key columns ----
        CFrag r0, r1;
        for (int i = 0; i < 8; ++i) { r0.f[i] = 0.0f; r1.f[i] = 0.0f; }
        {
            Frag bk;
            int kx = kxmin + lm; if (kx > 63) kx = 63;
            const _Float16* kp = Kh + (size_t)(krowbase + kx) * 256 + h * 32 + hi * 16;
            bk.q4[0] = ((const B16*)kp)[0];
            bk.q4[1] = ((const B16*)kp)[1];
            r0.v = __builtin_amdgcn_wmma_f32_16x16x32_f16(
                false, aq.v, false, bk.v, (short)0, r0.v, false, false);

            kx = kxmin + 16 + lm; if (kx > 63) kx = 63;
            kp = Kh + (size_t)(krowbase + kx) * 256 + h * 32 + hi * 16;
            bk.q4[0] = ((const B16*)kp)[0];
            bk.q4[1] = ((const B16*)kp)[1];
            r1.v = __builtin_amdgcn_wmma_f32_16x16x32_f16(
                false, aq.v, false, bk.v, (short)0, r1.v, false, false);
        }
        for (int r = 0; r < 8; ++r) {
            Rtile[h][r + hi * 8][lm]      = r0.f[r];
            Rtile[h][r + hi * 8][16 + lm] = r1.f[r];
        }
        __syncthreads();

        // ---- online softmax: 2 lanes per query, 7 scores per query ----
        const int q = L >> 1, halfo = L & 1;
        const int cxq = clampi(x0 + q, 3, 60);
        const int j0 = cxq - 3 - kxmin;            // in [0, width-7]
        const int sbase = (dy + 3) * 7;
        const int i0 = halfo ? 4 : 0;
        const int ni = halfo ? 3 : 4;
        float sc[4];
        float mx = -3e38f;
        for (int i = 0; i < ni; ++i) {
            int ii = i0 + i;
            float v = (Rtile[h][q][j0 + ii] + (float)S49t[h][q][sbase + ii]) * scale;
            sc[i] = v;
            mx = fmaxf(mx, v);
        }
        mx = fmaxf(mx, __shfl_xor(mx, 1, 32));
        float mo = mbuf[h][q];
        float mn = fmaxf(mo, mx);
        float alpha = __expf(mo - mn);
        float sum = 0.0f;
        float pv[4];
        for (int i = 0; i < ni; ++i) { pv[i] = __expf(sc[i] - mn); sum += pv[i]; }
        sum += __shfl_xor(sum, 1, 32);

        {   // zero my half of the P row, then scatter the 7 probabilities
            B16 z; z.w[0] = z.w[1] = z.w[2] = z.w[3] = 0;
            ((B16*)&Ptile[h][q][halfo * 16])[0] = z;
            ((B16*)&Ptile[h][q][halfo * 16])[1] = z;
        }
        __builtin_amdgcn_wave_barrier();
        for (int i = 0; i < ni; ++i)
            Ptile[h][q][j0 + i0 + i] = (_Float16)pv[i];
        if (halfo == 0) {
            mbuf[h][q] = mn;
            lbuf[h][q] = lbuf[h][q] * alpha + sum;
            abuf[h][q] = alpha;
        }
        __syncthreads();

        // ---- P A-fragment + acc rescale + P@V ----
        Frag ap;
        for (int v = 0; v < 8; ++v) {
            int kk = 2 * v + hi * 16;
            ap.u[v] = *(const unsigned*)&Ptile[h][lm][kk];
        }
        for (int r = 0; r < 8; ++r) {
            float a8 = abuf[h][r + hi * 8];
            acc0.f[r] *= a8;
            acc1.f[r] *= a8;
        }
        Frag bv0, bv1;
        for (int jj = 0; jj < 16; ++jj) {
            int krow = hi * 16 + jj;               // key index (rows >= width are 0)
            bv0.e[jj] = Vtile[krow][h * 32 + lm];
            bv1.e[jj] = Vtile[krow][h * 32 + 16 + lm];
        }
        acc0.v = __builtin_amdgcn_wmma_f32_16x16x32_f16(
            false, ap.v, false, bv0.v, (short)0, acc0.v, false, false);
        acc1.v = __builtin_amdgcn_wmma_f32_16x16x32_f16(
            false, ap.v, false, bv1.v, (short)0, acc1.v, false, false);
    }

    // ---- epilogue: out = acc / (l + 1e-8), f16 for the Wo GEMM ----
    for (int r = 0; r < 8; ++r) {
        int q = r + hi * 8;
        float inv = 1.0f / (lbuf[h][q] + 1e-8f);
        size_t base = (size_t)(base_row + q) * 256 + h * 32;
        Oh[base + lm]      = (_Float16)(acc0.f[r] * inv);
        Oh[base + 16 + lm] = (_Float16)(acc1.f[r] * inv);
    }
}

// ---------------------------------------------------------------------------
// Launch
// ---------------------------------------------------------------------------
extern "C" void kernel_launch(void* const* d_in, const int* in_sizes, int n_in,
                              void* d_out, int out_size, void* d_ws, size_t ws_size,
                              hipStream_t stream) {
    const float* x   = (const float*)d_in[0];
    const float* Wq  = (const float*)d_in[1];
    const float* Wk  = (const float*)d_in[2];
    const float* Wv  = (const float*)d_in[3];
    const float* Wo  = (const float*)d_in[4];
    const float* pos = (const float*)d_in[5];
    float* out = (float*)d_out;

    char* ws = (char*)d_ws;
    const size_t MB8 = (size_t)16384 * 256 * sizeof(_Float16);  // 8 MB
    _Float16* xh    = (_Float16*)(ws);
    _Float16* Qh    = (_Float16*)(ws + 1 * MB8);
    _Float16* Kh    = (_Float16*)(ws + 2 * MB8);
    _Float16* Vh    = (_Float16*)(ws + 3 * MB8);
    _Float16* Oh    = (_Float16*)(ws + 4 * MB8);
    _Float16* WqkvT = (_Float16*)(ws + 5 * MB8);
    _Float16* WoT   = (_Float16*)(ws + 5 * MB8 + (size_t)NWQKV * 2);
    _Float16* posh  = (_Float16*)(ws + 5 * MB8 + (size_t)(NWQKV + NWO) * 2);

    // 1) convert / pack
    {
        int total = NX_ELEMS + NWQKV + NWO + NPOS;
        int grid = (total + 255) / 256;
        lsa_convert_pack<<<grid, 256, 0, stream>>>(x, Wq, Wk, Wv, Wo, pos,
                                                   xh, WqkvT, WoT, posh);
    }
    // 2) fused QKV projection (+ query pos-emb), N = 768
    lsa_wmma_gemm<<<(16384 / 128) * (768 / 64), 256, 0, stream>>>(
        xh, WqkvT, Qh, Kh, Vh, pos, nullptr, 768, 0);
    // 3) windowed flash attention
    lsa_attention<<<4 * 64 * 4, 256, 0, stream>>>(Qh, Kh, Vh, posh, Oh);
    // 4) output projection, N = 256, f32 result
    lsa_wmma_gemm<<<(16384 / 128) * (256 / 64), 256, 0, stream>>>(
        Oh, WoT, nullptr, nullptr, nullptr, nullptr, out, 256, 1);
}
